// CustomLSTM_13486197309806
// MI455X (gfx1250) — matrix operand order repaired and checked by reading, not compiled, not run on hardware
//
#include <hip/hip_runtime.h>

// ---------------- problem constants ----------------
constexpr int B = 32, S = 512, I = 1024, H = 1024;
constexpr int G4H = 4 * H;
constexpr int KC  = 256;            // K chunk staged in LDS per K-half per iteration
constexpr int KHALF = H / 2;        // 512: K range owned by each wave group

// ---------------- vector types for WMMA ----------------
typedef __attribute__((ext_vector_type(16))) __bf16 v16bf;
typedef __attribute__((ext_vector_type(8)))  __bf16 v8bf;
typedef __attribute__((ext_vector_type(4)))  __bf16 v4bf;
typedef __attribute__((ext_vector_type(8)))  float  v8f;

// ---------------- workspace layout (bytes) ----------------
constexpr size_t SZ_XBF = (size_t)B * S * I * 2;       // x in bf16
constexpr size_t SZ_W   = (size_t)G4H * I * 2;         // one weight matrix bf16 (I==H)
constexpr size_t OFF_XBF = 0;
constexpr size_t OFF_W0I = OFF_XBF + SZ_XBF;
constexpr size_t OFF_W0H = OFF_W0I + SZ_W;
constexpr size_t OFF_W1I = OFF_W0H + SZ_W;
constexpr size_t OFF_W1H = OFF_W1I + SZ_W;
constexpr size_t OFF_B0  = OFF_W1H + SZ_W;
constexpr size_t OFF_B1  = OFF_B0 + (size_t)G4H * 4;
constexpr size_t OFF_H0  = OFF_B1 + (size_t)G4H * 4;               // 2 buffers
constexpr size_t OFF_H1  = OFF_H0 + 2 * (size_t)B * H * 4;         // 2 buffers
constexpr size_t OFF_C0  = OFF_H1 + 2 * (size_t)B * H * 4;
constexpr size_t OFF_C1  = OFF_C0 + (size_t)B * H * 4;

// ---------------- f32 -> bf16 bulk convert ----------------
__global__ void k_f32_to_bf16(const float* __restrict__ src, __bf16* __restrict__ dst, int n4) {
  int i = blockIdx.x * blockDim.x + threadIdx.x;
  if (i >= n4) return;
  const float4 v = ((const float4*)src)[i];
  v4bf o = {(__bf16)v.x, (__bf16)v.y, (__bf16)v.z, (__bf16)v.w};
  ((v4bf*)dst)[i] = o;
}

// ---------------- fold biases, zero states ----------------
__global__ void k_prep(const float* __restrict__ bih0, const float* __restrict__ bhh0,
                       const float* __restrict__ bih1, const float* __restrict__ bhh1,
                       float* __restrict__ bias0, float* __restrict__ bias1,
                       float* __restrict__ h0, float* __restrict__ h1,
                       float* __restrict__ c0, float* __restrict__ c1) {
  int i = blockIdx.x * blockDim.x + threadIdx.x;
  if (i < G4H) { bias0[i] = bih0[i] + bhh0[i]; bias1[i] = bih1[i] + bhh1[i]; }
  if (i < 2 * B * H) { h0[i] = 0.f; h1[i] = 0.f; }
  if (i < B * H)     { c0[i] = 0.f; c1[i] = 0.f; }
}

__device__ __forceinline__ float sigmoidf_fast(float x) {
  return 1.0f / (1.0f + __expf(-x));
}

// ---------------- one LSTM cell step for one layer ----------------
// grid = (H/16, B/16), block = 256 (8 waves).
// Wave w: gate = w&3 (i,f,g,o), K-half = w>>2. Each wave accumulates its gate's
// 16x16 tile over 512 K values with v_wmma_f32_16x16x32_bf16, keeping the
// input-GEMM and recurrent-GEMM partial sums in SEPARATE accumulators so
// adjacent WMMAs are independent (2 chains of 16 instead of 1 chain of 32).
// K-half partials + gate exchange are reduced through LDS, then the
// sigmoid/tanh/c/h pointwise update runs in the same kernel.
__global__ __launch_bounds__(256) void k_lstm_step(
    const __bf16* __restrict__ xbf,     // layer0: bf16 x base (indexed by t); else nullptr
    const float*  __restrict__ hin_l,   // layer1: f32 [B,H] lower-layer h for this step; else nullptr
    const __bf16* __restrict__ Wi,      // [4H, 1024] bf16 row-major
    const __bf16* __restrict__ Wh,      // [4H, 1024] bf16 row-major
    const float*  __restrict__ bias,    // [4H] (bih + bhh)
    const float*  __restrict__ h_prev,  // [B,H] recurrent input
    float*        __restrict__ h_next,  // [B,H] recurrent output (double-buffered)
    float*        __restrict__ c_state, // [B,H] cell state (block-private per element)
    float*        __restrict__ seq_out, // layer1: d_out [B,S,H]; else nullptr
    int t) {
  __shared__ __bf16 a_in[2][16][KC];    // layer-input A tile, per K-half
  __shared__ __bf16 a_h [2][16][KC];    // recurrent-h A tile, per K-half
  __shared__ float  gacc[8][16][16];    // per-(gate,K-half) accumulator exchange

  const int tid  = threadIdx.x;
  const int wave = tid >> 5;
  const int gate = wave & 3;            // 0=i 1=f 2=g 3=o
  const int ks   = wave >> 2;           // K-half owned by this wave
  const int lane = tid & 31;
  const int ub   = blockIdx.x;          // hidden-unit tile (16 units)
  const int rb   = blockIdx.y;          // batch-row tile (16 rows)

  v8f acc_x = {};                       // input-GEMM partial
  v8f acc_h = {};                       // recurrent-GEMM partial

  // Per-lane fragment geometry (CDNA5 wave32 WMMA layouts):
  // A 16x32 bf16: lane m / m+16 hold row m; elems 0..7 => K {0..7|8..15}, elems 8..15 => +16.
  // B 32x16 bf16: lane holds col n = lane&15; same K order as A: elems 0..7 => K {0..7|8..15}, elems 8..15 => +16.
  const int r     = lane & 15;
  const int half8 = (lane >> 4) << 3;
  const int n     = lane & 15;
  const int kh    = (lane >> 4) << 3;
  const __bf16* wi_row = Wi + ((size_t)gate * H + (size_t)ub * 16 + n) * (size_t)H
                            + (size_t)ks * KHALF + kh;
  const __bf16* wh_row = Wh + ((size_t)gate * H + (size_t)ub * 16 + n) * (size_t)H
                            + (size_t)ks * KHALF + kh;

  constexpr int CH4 = KC / 4;           // 4-element groups per row per chunk
  for (int kc = 0; kc < KHALF; kc += KC) {
    // ---- cooperative LDS staging of both K-halves' A tiles (f32->bf16 on the fly) ----
    for (int e = tid; e < 2 * 16 * CH4; e += 256) {
      const int hf  = e / (16 * CH4);                  // which K-half
      const int rem = e - hf * 16 * CH4;
      const int rr  = rem / CH4;
      const int c4  = (rem % CH4) * 4;
      const int gk  = hf * KHALF + kc + c4;            // global K offset
      const int gr  = rb * 16 + rr;                    // global batch row
      if (xbf) {                                       // layer0: x[b, t, k] already bf16
        *(v4bf*)&a_in[hf][rr][c4] =
            *(const v4bf*)(xbf + (size_t)gr * ((size_t)S * I) + (size_t)t * I + gk);
      } else {                                         // layer1: lower-layer h, f32
        const float4 v = *(const float4*)(hin_l + (size_t)gr * H + gk);
        v4bf o = {(__bf16)v.x, (__bf16)v.y, (__bf16)v.z, (__bf16)v.w};
        *(v4bf*)&a_in[hf][rr][c4] = o;
      }
      {
        const float4 v = *(const float4*)(h_prev + (size_t)gr * H + gk);
        v4bf o = {(__bf16)v.x, (__bf16)v.y, (__bf16)v.z, (__bf16)v.w};
        *(v4bf*)&a_h[hf][rr][c4] = o;
      }
    }
    __syncthreads();

    if (kc == 0) {                                     // pull next weight chunk toward L2/L0
      __builtin_prefetch(wi_row + KC, 0, 1);
      __builtin_prefetch(wh_row + KC, 0, 1);
    }

    // ---- fused input + recurrent GEMM over this wave's K chunk ----
#pragma unroll
    for (int kk = 0; kk < KC; kk += 32) {
      v8bf alo = *(const v8bf*)&a_in[ks][r][kk + half8];
      v8bf ahi = *(const v8bf*)&a_in[ks][r][kk + half8 + 16];
      v16bf afrag = __builtin_shufflevector(alo, ahi, 0,1,2,3,4,5,6,7,8,9,10,11,12,13,14,15);
      v8bf blo = *(const v8bf*)(wi_row + kc + kk);
      v8bf bhi = *(const v8bf*)(wi_row + kc + kk + 16);
      v16bf bfrag = __builtin_shufflevector(blo, bhi, 0,1,2,3,4,5,6,7,8,9,10,11,12,13,14,15);
      acc_x = __builtin_amdgcn_wmma_f32_16x16x32_bf16(false, afrag, false, bfrag,
                                                      (short)0, acc_x, false, false);

      v8bf hlo = *(const v8bf*)&a_h[ks][r][kk + half8];
      v8bf hhi = *(const v8bf*)&a_h[ks][r][kk + half8 + 16];
      v16bf hfrag = __builtin_shufflevector(hlo, hhi, 0,1,2,3,4,5,6,7,8,9,10,11,12,13,14,15);
      v8bf clo = *(const v8bf*)(wh_row + kc + kk);
      v8bf chi = *(const v8bf*)(wh_row + kc + kk + 16);
      v16bf wfrag = __builtin_shufflevector(clo, chi, 0,1,2,3,4,5,6,7,8,9,10,11,12,13,14,15);
      acc_h = __builtin_amdgcn_wmma_f32_16x16x32_bf16(false, hfrag, false, wfrag,
                                                      (short)0, acc_h, false, false);
    }
    __syncthreads();                                   // LDS reused next chunk
  }

  // ---- exchange gate tiles through LDS (D layout: elem e -> M=(lane>>4)*8+e, N=lane&15) ----
  const v8f tot = acc_x + acc_h;
#pragma unroll
  for (int e = 0; e < 8; ++e)
    gacc[wave][((lane >> 4) << 3) + e][lane & 15] = tot[e];
  __syncthreads();

  // ---- pointwise LSTM update: 256 (m,n) elements over 256 threads ----
  {
    const int m  = tid >> 4;
    const int nn = tid & 15;
    const int u  = ub * 16 + nn;                       // hidden unit
    const int b  = rb * 16 + m;                        // batch row
    const float gi = gacc[0][m][nn] + gacc[4][m][nn] + bias[0 * H + u];
    const float gf = gacc[1][m][nn] + gacc[5][m][nn] + bias[1 * H + u];
    const float gg = gacc[2][m][nn] + gacc[6][m][nn] + bias[2 * H + u];
    const float go = gacc[3][m][nn] + gacc[7][m][nn] + bias[3 * H + u];
    const float si = sigmoidf_fast(gi);
    const float sf = sigmoidf_fast(gf);
    const float tg = tanhf(gg);
    const float so = sigmoidf_fast(go);
    const size_t ix = (size_t)b * H + u;
    const float cn = sf * c_state[ix] + si * tg;
    const float hn = so * tanhf(cn);
    c_state[ix] = cn;
    h_next[ix]  = hn;
    if (seq_out) seq_out[(size_t)b * ((size_t)S * H) + (size_t)t * H + u] = hn;
  }
}

// ---------------- copy final h_n / c_n into d_out tail ----------------
__global__ void k_finalize(const float* __restrict__ h0, const float* __restrict__ h1,
                           const float* __restrict__ c0, const float* __restrict__ c1,
                           float* __restrict__ out) {
  const int i = blockIdx.x * blockDim.x + threadIdx.x;
  if (i >= B * H) return;
  const size_t base = (size_t)B * S * H;
  out[base + i]                       = h0[i];
  out[base + (size_t)B * H + i]       = h1[i];
  out[base + 2 * (size_t)B * H + i]   = c0[i];
  out[base + 3 * (size_t)B * H + i]   = c1[i];
}

extern "C" void kernel_launch(void* const* d_in, const int* in_sizes, int n_in,
                              void* d_out, int out_size, void* d_ws, size_t ws_size,
                              hipStream_t stream) {
  (void)in_sizes; (void)n_in; (void)out_size; (void)ws_size;
  const float* x    = (const float*)d_in[0];
  const float* Wih0 = (const float*)d_in[1];
  const float* Whh0 = (const float*)d_in[2];
  const float* bih0 = (const float*)d_in[3];
  const float* bhh0 = (const float*)d_in[4];
  const float* Wih1 = (const float*)d_in[5];
  const float* Whh1 = (const float*)d_in[6];
  const float* bih1 = (const float*)d_in[7];
  const float* bhh1 = (const float*)d_in[8];
  float* out = (float*)d_out;

  char* ws = (char*)d_ws;
  __bf16* xbf = (__bf16*)(ws + OFF_XBF);
  __bf16* w0i = (__bf16*)(ws + OFF_W0I);
  __bf16* w0h = (__bf16*)(ws + OFF_W0H);
  __bf16* w1i = (__bf16*)(ws + OFF_W1I);
  __bf16* w1h = (__bf16*)(ws + OFF_W1H);
  float* bias0 = (float*)(ws + OFF_B0);
  float* bias1 = (float*)(ws + OFF_B1);
  float* h0    = (float*)(ws + OFF_H0);   // 2 ping-pong buffers
  float* h1    = (float*)(ws + OFF_H1);   // 2 ping-pong buffers
  float* c0    = (float*)(ws + OFF_C0);
  float* c1    = (float*)(ws + OFF_C1);

  // Prep: bf16 conversions + bias fold + state zeroing (every call; deterministic)
  const int xn4 = B * S * I / 4;
  const int wn4 = G4H * I / 4;
  k_f32_to_bf16<<<(xn4 + 255) / 256, 256, 0, stream>>>(x, xbf, xn4);
  k_f32_to_bf16<<<(wn4 + 255) / 256, 256, 0, stream>>>(Wih0, w0i, wn4);
  k_f32_to_bf16<<<(wn4 + 255) / 256, 256, 0, stream>>>(Whh0, w0h, wn4);
  k_f32_to_bf16<<<(wn4 + 255) / 256, 256, 0, stream>>>(Wih1, w1i, wn4);
  k_f32_to_bf16<<<(wn4 + 255) / 256, 256, 0, stream>>>(Whh1, w1h, wn4);
  k_prep<<<(2 * B * H + 255) / 256, 256, 0, stream>>>(bih0, bhh0, bih1, bhh1,
                                                      bias0, bias1, h0, h1, c0, c1);

  // Recurrence: one launch per (step, layer); h double-buffered.
  const dim3 grid(H / 16, B / 16);
  for (int t = 0; t < S; ++t) {
    const int pi = t & 1, po = pi ^ 1;
    k_lstm_step<<<grid, 256, 0, stream>>>(xbf, nullptr, w0i, w0h, bias0,
                                          h0 + (size_t)pi * B * H, h0 + (size_t)po * B * H,
                                          c0, nullptr, t);
    k_lstm_step<<<grid, 256, 0, stream>>>(nullptr, h0 + (size_t)po * B * H, w1i, w1h, bias1,
                                          h1 + (size_t)pi * B * H, h1 + (size_t)po * B * H,
                                          c1, out, t);
  }
  // S is even: final states sit in buffer 0 of each ping-pong pair.
  k_finalize<<<(B * H + 255) / 256, 256, 0, stream>>>(h0, h1, c0, c1, out);
}
